// ImageTextMatchingModel_78340203479445
// MI455X (gfx1250) — matrix operand order repaired and checked by reading, not compiled, not run on hardware
//
#include <hip/hip_runtime.h>
#include <hip/hip_bf16.h>

// ---------------------------------------------------------------------------
// Types / WMMA helpers (CDNA5 gfx1250, wave32)
// ---------------------------------------------------------------------------
typedef _Float16 h16 __attribute__((ext_vector_type(16)));
typedef _Float16 h8  __attribute__((ext_vector_type(8)));
typedef float    f8  __attribute__((ext_vector_type(8)));

__device__ __forceinline__ h16 cat16(h8 lo, h8 hi_) {
  return __builtin_shufflevector(lo, hi_, 0,1,2,3,4,5,6,7,8,9,10,11,12,13,14,15);
}
__device__ __forceinline__ f8 wmma_f16(h16 a, h16 b, f8 c) {
  // D = A(16x32 f16) * B(32x16 f16) + C(16x16 f32)
  return __builtin_amdgcn_wmma_f32_16x16x32_f16(false, a, false, b, (short)0, c, false, false);
}

// Problem constants
#define BB     64
#define CIMG   1024
#define NPATCH 196
#define TLEN   77
#define CTXT   1024
#define DD     512
#define HH     8

// ---------------------------------------------------------------------------
// float32 -> float16 flat convert
// ---------------------------------------------------------------------------
__global__ void cvt_f32_f16(const float* __restrict__ src, _Float16* __restrict__ dst, int n) {
  int i = blockIdx.x * 256 + threadIdx.x;
  if (i < n) dst[i] = (_Float16)src[i];
}

// ---------------------------------------------------------------------------
// Weight transpose+convert: W[K,N] f32 -> WT[N,K] f16   (K,N multiples of 32)
// block (32,8)
// ---------------------------------------------------------------------------
__global__ void transpose_cvt_w(const float* __restrict__ src, _Float16* __restrict__ dst,
                                int K, int N) {
  __shared__ float t[32][33];
  int k0 = blockIdx.x * 32, n0 = blockIdx.y * 32;
  int tx = threadIdx.x, ty = threadIdx.y;
#pragma unroll
  for (int i = 0; i < 4; ++i)
    t[ty + i * 8][tx] = src[(size_t)(k0 + ty + i * 8) * N + (n0 + tx)];
  __syncthreads();
#pragma unroll
  for (int i = 0; i < 4; ++i)
    dst[(size_t)(n0 + ty + i * 8) * K + (k0 + tx)] = (_Float16)t[tx][ty + i * 8];
}

// ---------------------------------------------------------------------------
// Image transpose+convert: img[B,1024,196] f32 -> imgT[B*196,1024] f16
// grid (ceil(196/32)=7, 1024/32=32, B), block (32,8)
// ---------------------------------------------------------------------------
__global__ void transpose_cvt_img(const float* __restrict__ src, _Float16* __restrict__ dst) {
  __shared__ float t[32][33];
  int b = blockIdx.z;
  int n0 = blockIdx.x * 32, c0 = blockIdx.y * 32;
  int tx = threadIdx.x, ty = threadIdx.y;
#pragma unroll
  for (int i = 0; i < 4; ++i) {
    int c = c0 + ty + i * 8, n = n0 + tx;
    t[ty + i * 8][tx] = (n < NPATCH) ? src[((size_t)b * CIMG + c) * NPATCH + n] : 0.f;
  }
  __syncthreads();
#pragma unroll
  for (int i = 0; i < 4; ++i) {
    int n = n0 + ty + i * 8, c = c0 + tx;
    if (n < NPATCH)
      dst[((size_t)b * NPATCH + n) * CIMG + c] = (_Float16)t[tx][ty + i * 8];
  }
}

// ---------------------------------------------------------------------------
// Generic WMMA GEMM:  C[M,N] = A[M,K](f16) * BT[N,K]^T(f16) + bias[N]
// 128x128 tile / workgroup (256 thr = 8 waves); 32-wide K chunks,
// double-buffered LDS (one barrier per chunk, global loads overlap WMMA).
// ---------------------------------------------------------------------------
template <bool W32, bool W16>
__global__ __launch_bounds__(256) void gemm_wmma(
    const _Float16* __restrict__ A, const _Float16* __restrict__ BT,
    const float* __restrict__ bias, float* __restrict__ C32,
    _Float16* __restrict__ C16, int M, int K, int N) {
  constexpr int LDT = 40;  // 32 + 8 halves pad -> conflict-free b128 reads
  __shared__ alignas(16) _Float16 As[2][128 * LDT];
  __shared__ alignas(16) _Float16 Bs[2][128 * LDT];

  const int tid = threadIdx.x, lane = tid & 31, wv = tid >> 5;
  const int l16 = lane & 15, hiw = (lane >> 4) & 1;
  const int rowbase = blockIdx.y * 128;
  const int colbase = blockIdx.x * 128;
  const int wr = (wv & 3) * 32;   // wave row offset within tile
  const int wc = (wv >> 2) * 64;  // wave col offset within tile
  const int aoff = hiw ? 16 : 0;
  const int boff = hiw ? 16 : 0;

  auto load_tile = [&](int kc, int buf) {
    for (int i = tid; i < 512; i += 256) {
      int r = i >> 2, seg = i & 3;
      int gr = rowbase + r;
      uint4 v = make_uint4(0u, 0u, 0u, 0u);
      if (gr < M) v = *(const uint4*)(A + (size_t)gr * K + kc + seg * 8);
      *(uint4*)(As[buf] + r * LDT + seg * 8) = v;
    }
    for (int i = tid; i < 512; i += 256) {
      int r = i >> 2, seg = i & 3;
      uint4 v = *(const uint4*)(BT + (size_t)(colbase + r) * K + kc + seg * 8);
      *(uint4*)(Bs[buf] + r * LDT + seg * 8) = v;
    }
  };

  f8 acc[2][4] = {};
  const int nc = K >> 5;
  load_tile(0, 0);

  for (int c = 0; c < nc; ++c) {
    __syncthreads();
    if (c + 1 < nc) load_tile((c + 1) * 32, (c + 1) & 1);
    const int buf = c & 1;

    // Preload all fragments, then run the WMMA chain back-to-back.
    h16 a[2], b[4];
#pragma unroll
    for (int it = 0; it < 2; ++it) {
      const _Float16* p = As[buf] + (wr + it * 16 + l16) * LDT + aoff;
      a[it] = cat16(*(const h8*)p, *(const h8*)(p + 8));
    }
#pragma unroll
    for (int jt = 0; jt < 4; ++jt) {
      const _Float16* p = Bs[buf] + (wc + jt * 16 + l16) * LDT + boff;
      b[jt] = cat16(*(const h8*)p, *(const h8*)(p + 8));
    }
#pragma unroll
    for (int it = 0; it < 2; ++it)
#pragma unroll
      for (int jt = 0; jt < 4; ++jt)
        acc[it][jt] = wmma_f16(a[it], b[jt], acc[it][jt]);
  }

  // Epilogue: C layout -> VGPR r holds row (r + 8*hiw), col = l16
#pragma unroll
  for (int jt = 0; jt < 4; ++jt) {
    int col = colbase + wc + jt * 16 + l16;
    float bv = bias[col];
#pragma unroll
    for (int it = 0; it < 2; ++it) {
#pragma unroll
      for (int r = 0; r < 8; ++r) {
        int row = rowbase + wr + it * 16 + r + hiw * 8;
        if (row < M) {
          float v = acc[it][jt][r] + bv;
          if (W32) C32[(size_t)row * N + col] = v;
          if (W16) C16[(size_t)row * N + col] = (_Float16)v;
        }
      }
    }
  }
}

// ---------------------------------------------------------------------------
// Fused cross-attention per (b,h): scores = q k^T/8, softmax, attn = w v
// grid(512), block(256 = 8 waves)
// ---------------------------------------------------------------------------
__global__ __launch_bounds__(256) void attn_kernel(
    const _Float16* __restrict__ qH, const _Float16* __restrict__ kH,
    const _Float16* __restrict__ vH, float* __restrict__ w_buf,
    _Float16* __restrict__ attnH) {
  constexpr int LQW = 96;  // q rows (64 halves) reused as w rows (96 halves)
  constexpr int LK = 64;
  constexpr int LV = 96;
  __shared__ alignas(16) _Float16 qw[208 * LQW];
  __shared__ alignas(16) _Float16 ks[80 * LK];
  __shared__ alignas(16) _Float16 vs[64 * LV];

  const int bh = blockIdx.x, b = bh >> 3, h = bh & 7;
  const int tid = threadIdx.x, lane = tid & 31, wv = tid >> 5;
  const int l16 = lane & 15, hiw = (lane >> 4) & 1;
  const int aoff = hiw ? 16 : 0;
  const int boff = hiw ? 16 : 0;

  // Phase 0: zero LDS (pads must be zero)
  uint4 z4 = make_uint4(0u, 0u, 0u, 0u);
  for (int i = tid; i < 208 * LQW / 8; i += 256) ((uint4*)qw)[i] = z4;
  for (int i = tid; i < 80 * LK / 8; i += 256) ((uint4*)ks)[i] = z4;
  for (int i = tid; i < 64 * LV / 8; i += 256) ((uint4*)vs)[i] = z4;
  __syncthreads();

  // Phase 1: stage q (196x64), k (77x64), v transposed (64 x 77)
  const _Float16* qb = qH + (size_t)b * NPATCH * DD + h * 64;
  const _Float16* kb = kH + (size_t)b * TLEN * DD + h * 64;
  const _Float16* vb = vH + (size_t)b * TLEN * DD + h * 64;
  for (int i = tid; i < NPATCH * 8; i += 256) {
    int r = i >> 3, seg = i & 7;
    *(uint4*)(qw + r * LQW + seg * 8) = *(const uint4*)(qb + (size_t)r * DD + seg * 8);
  }
  for (int i = tid; i < TLEN * 8; i += 256) {
    int r = i >> 3, seg = i & 7;
    *(uint4*)(ks + r * LK + seg * 8) = *(const uint4*)(kb + (size_t)r * DD + seg * 8);
  }
  for (int i = tid; i < TLEN * 32; i += 256) {
    int t = i >> 5, c = (i & 31) * 2;
    const _Float16* p = vb + (size_t)t * DD + c;
    vs[(c + 0) * LV + t] = p[0];
    vs[(c + 1) * LV + t] = p[1];
  }
  __syncthreads();

  // 13 row-strips of 16; wave wv owns strips wv and wv+8 (uniform per wave)
  for (int half = 0; half < 2; ++half) {
    int rt = wv + half * 8;
    bool active = rt < 13;  // wave-uniform -> EXEC stays all-1 around WMMA
    if (active) {
      // ---- scores strip: (16 x 64) @ (64 x 80) ----
      f8 acc[5] = {};
#pragma unroll
      for (int kc = 0; kc < 64; kc += 32) {
        const _Float16* ap = qw + (rt * 16 + l16) * LQW + kc + aoff;
        h16 a = cat16(*(const h8*)ap, *(const h8*)(ap + 8));
        h16 bb[5];
#pragma unroll
        for (int j = 0; j < 5; ++j) {
          const _Float16* bp = ks + (j * 16 + l16) * LK + kc + boff;
          bb[j] = cat16(*(const h8*)bp, *(const h8*)(bp + 8));
        }
#pragma unroll
        for (int j = 0; j < 5; ++j) acc[j] = wmma_f16(a, bb[j], acc[j]);
      }
      // ---- softmax over t=0..76 (cols 77..79 masked), rows in C layout ----
      bool inval = (l16 > 12);  // j==4 -> col = 64+l16 >= 77
      f8 m;
#pragma unroll
      for (int r = 0; r < 8; ++r) {
        float mm = -1e30f;
#pragma unroll
        for (int j = 0; j < 5; ++j) {
          float v = acc[j][r] * 0.125f;  // 1/sqrt(hd=64)
          acc[j][r] = v;
          if (j < 4 || !inval) mm = fmaxf(mm, v);
        }
        m[r] = mm;
      }
#pragma unroll
      for (int s = 1; s < 16; s <<= 1)
#pragma unroll
        for (int r = 0; r < 8; ++r) m[r] = fmaxf(m[r], __shfl_xor(m[r], s, 32));
      f8 e[5], sum;
#pragma unroll
      for (int r = 0; r < 8; ++r) {
        float s0 = 0.f;
#pragma unroll
        for (int j = 0; j < 5; ++j) {
          float ev = ((j == 4) && inval) ? 0.f : __expf(acc[j][r] - m[r]);
          e[j][r] = ev;
          s0 += ev;
        }
        sum[r] = s0;
      }
#pragma unroll
      for (int s = 1; s < 16; s <<= 1)
#pragma unroll
        for (int r = 0; r < 8; ++r) sum[r] += __shfl_xor(sum[r], s, 32);
      f8 rs;
#pragma unroll
      for (int r = 0; r < 8; ++r) rs[r] = 1.f / sum[r];
      // write w: f16 into own strip rows of qw (q rows now dead), f32 to global
#pragma unroll
      for (int j = 0; j < 5; ++j) {
        int col = j * 16 + l16;
#pragma unroll
        for (int r = 0; r < 8; ++r) {
          int row = rt * 16 + r + hiw * 8;
          float wval = e[j][r] * rs[r];
          qw[row * LQW + col] = (_Float16)wval;
          if (col < TLEN && row < NPATCH)
            w_buf[((size_t)bh * NPATCH + row) * TLEN + col] = wval;
        }
      }
    }
    __syncthreads();
    if (active) {
      // ---- attn strip: (16 x 96) @ (96 x 64), K padded with zeros ----
      f8 acc2[4] = {};
#pragma unroll
      for (int kc = 0; kc < 96; kc += 32) {
        const _Float16* ap = qw + (rt * 16 + l16) * LQW + kc + aoff;
        h16 a = cat16(*(const h8*)ap, *(const h8*)(ap + 8));
        h16 bb[4];
#pragma unroll
        for (int j = 0; j < 4; ++j) {
          const _Float16* bp = vs + (j * 16 + l16) * LV + kc + boff;
          bb[j] = cat16(*(const h8*)bp, *(const h8*)(bp + 8));
        }
#pragma unroll
        for (int j = 0; j < 4; ++j) acc2[j] = wmma_f16(a, bb[j], acc2[j]);
      }
      _Float16* ob = attnH + (size_t)b * NPATCH * DD + h * 64;
#pragma unroll
      for (int j = 0; j < 4; ++j) {
        int col = j * 16 + l16;
#pragma unroll
        for (int r = 0; r < 8; ++r) {
          int row = rt * 16 + r + hiw * 8;
          if (row < NPATCH) ob[(size_t)row * DD + col] = (_Float16)acc2[j][r];
        }
      }
    }
    __syncthreads();
  }
}

// ---------------------------------------------------------------------------
// attn_output_weights = mean over heads of w_buf
// ---------------------------------------------------------------------------
__global__ void mean_heads(const float* __restrict__ w_buf, float* __restrict__ out) {
  int i = blockIdx.x * 256 + threadIdx.x;
  const int per = NPATCH * TLEN;
  if (i >= BB * per) return;
  int b = i / per, rem = i - b * per;
  float s = 0.f;
#pragma unroll
  for (int h = 0; h < HH; ++h) s += w_buf[(size_t)(b * HH + h) * per + rem];
  out[i] = s * (1.f / HH);
}

// out[b*512+d] = mean_r in[(b*rows + r)*512 + d]
__global__ void mean_rows(const float* __restrict__ in, float* __restrict__ out,
                          int rows, float inv) {
  int i = blockIdx.x * 256 + threadIdx.x;
  if (i >= BB * DD) return;
  int b = i >> 9, d = i & 511;
  const float* p = in + (size_t)b * rows * DD + d;
  float s = 0.f;
  for (int r = 0; r < rows; ++r) s += p[(size_t)r * DD];
  out[i] = s * inv;
}

// score[i][j] = <avg_i, cls_j> / (max(||avg_i||,eps) * max(||cls_j||,eps))
__global__ void cosine_scores(const float* __restrict__ avg, const float* __restrict__ cls,
                              float* __restrict__ out) {
  int i = blockIdx.x * 256 + threadIdx.x;
  if (i >= BB * BB) return;
  int ri = i >> 6, rj = i & 63;
  const float* a = avg + (size_t)ri * DD;
  const float* c = cls + (size_t)rj * DD;
  float dot = 0.f, na = 0.f, nc = 0.f;
  for (int d = 0; d < DD; ++d) {
    float x = a[d], y = c[d];
    dot += x * y;
    na += x * x;
    nc += y * y;
  }
  float den = fmaxf(sqrtf(na), 1e-8f) * fmaxf(sqrtf(nc), 1e-8f);
  out[i] = dot / den;
}

// ---------------------------------------------------------------------------
// Launcher
// ---------------------------------------------------------------------------
extern "C" void kernel_launch(void* const* d_in, const int* in_sizes, int n_in,
                              void* d_out, int out_size, void* d_ws, size_t ws_size,
                              hipStream_t stream) {
  const float* image = (const float*)d_in[0];
  const float* text  = (const float*)d_in[1];
  const float* Wi = (const float*)d_in[2];  const float* bi = (const float*)d_in[3];
  const float* Wt = (const float*)d_in[4];  const float* bt = (const float*)d_in[5];
  const float* Wq = (const float*)d_in[6];  const float* bq = (const float*)d_in[7];
  const float* Wk = (const float*)d_in[8];  const float* bk = (const float*)d_in[9];
  const float* Wv = (const float*)d_in[10]; const float* bv = (const float*)d_in[11];
  const float* Wo = (const float*)d_in[12]; const float* bo = (const float*)d_in[13];

  float* out = (float*)d_out;
  float* out_score = out;                                  // 64*64
  float* out_attn  = out_score + 64 * 64;                  // 64*196*512
  float* out_w     = out_attn + (size_t)BB * NPATCH * DD;  // 64*196*77
  float* out_avg   = out_w + (size_t)BB * NPATCH * TLEN;   // 64*512
  float* out_cls   = out_avg + BB * DD;                    // 64*512
  float* out_ptxt  = out_cls + BB * DD;                    // 64*77*512

  char* wsb = (char*)d_ws;
  size_t off = 0;
  auto alloc = [&](size_t bytes) -> char* {
    char* p = wsb + off;
    off += (bytes + 255) & ~(size_t)255;
    return p;
  };
  const size_t MI = (size_t)BB * NPATCH;  // 12544
  const size_t MT = (size_t)BB * TLEN;    // 4928
  _Float16* imgTh = (_Float16*)alloc(MI * CIMG * 2);
  _Float16* textH = (_Float16*)alloc(MT * CTXT * 2);
  _Float16* WiT   = (_Float16*)alloc((size_t)CIMG * DD * 2);
  _Float16* WtT   = (_Float16*)alloc((size_t)CTXT * DD * 2);
  _Float16* WqT   = (_Float16*)alloc((size_t)DD * DD * 2);
  _Float16* WkT   = (_Float16*)alloc((size_t)DD * DD * 2);
  _Float16* WvT   = (_Float16*)alloc((size_t)DD * DD * 2);
  _Float16* WoT   = (_Float16*)alloc((size_t)DD * DD * 2);
  _Float16* ptxtH = (_Float16*)alloc(MT * DD * 2);
  _Float16* pimgH = (_Float16*)alloc(MI * DD * 2);
  _Float16* qHp   = (_Float16*)alloc(MI * DD * 2);
  _Float16* kHp   = (_Float16*)alloc(MT * DD * 2);
  _Float16* vHp   = (_Float16*)alloc(MT * DD * 2);
  _Float16* attnH = (_Float16*)alloc(MI * DD * 2);
  float*    w_buf = (float*)alloc((size_t)BB * HH * NPATCH * TLEN * 4);
  (void)ws_size; (void)in_sizes; (void)n_in; (void)out_size;

  dim3 b328(32, 8);
  // Weight transpose+convert (W[K,N] f32 -> WT[N,K] f16)
  transpose_cvt_w<<<dim3(CIMG / 32, DD / 32), b328, 0, stream>>>(Wi, WiT, CIMG, DD);
  transpose_cvt_w<<<dim3(CTXT / 32, DD / 32), b328, 0, stream>>>(Wt, WtT, CTXT, DD);
  transpose_cvt_w<<<dim3(DD / 32, DD / 32), b328, 0, stream>>>(Wq, WqT, DD, DD);
  transpose_cvt_w<<<dim3(DD / 32, DD / 32), b328, 0, stream>>>(Wk, WkT, DD, DD);
  transpose_cvt_w<<<dim3(DD / 32, DD / 32), b328, 0, stream>>>(Wv, WvT, DD, DD);
  transpose_cvt_w<<<dim3(DD / 32, DD / 32), b328, 0, stream>>>(Wo, WoT, DD, DD);
  // Activations -> f16
  transpose_cvt_img<<<dim3(7, CIMG / 32, BB), b328, 0, stream>>>(image, imgTh);
  {
    int n = (int)(MT * CTXT);
    cvt_f32_f16<<<(n + 255) / 256, 256, 0, stream>>>(text, textH, n);
  }

  // Projections (WMMA GEMMs)
  gemm_wmma<true, true><<<dim3(DD / 128, (MT + 127) / 128), 256, 0, stream>>>(
      textH, WtT, bt, out_ptxt, ptxtH, (int)MT, CTXT, DD);
  gemm_wmma<false, true><<<dim3(DD / 128, (MI + 127) / 128), 256, 0, stream>>>(
      imgTh, WiT, bi, nullptr, pimgH, (int)MI, CIMG, DD);
  gemm_wmma<false, true><<<dim3(DD / 128, (MI + 127) / 128), 256, 0, stream>>>(
      pimgH, WqT, bq, nullptr, qHp, (int)MI, DD, DD);
  gemm_wmma<false, true><<<dim3(DD / 128, (MT + 127) / 128), 256, 0, stream>>>(
      ptxtH, WkT, bk, nullptr, kHp, (int)MT, DD, DD);
  gemm_wmma<false, true><<<dim3(DD / 128, (MT + 127) / 128), 256, 0, stream>>>(
      ptxtH, WvT, bv, nullptr, vHp, (int)MT, DD, DD);

  // Fused attention per (b,h)
  attn_kernel<<<dim3(BB * HH), 256, 0, stream>>>(qHp, kHp, vHp, w_buf, attnH);

  // Output projection
  gemm_wmma<true, false><<<dim3(DD / 128, (MI + 127) / 128), 256, 0, stream>>>(
      attnH, WoT, bo, out_attn, nullptr, (int)MI, DD, DD);

  // Reductions
  {
    int n = BB * NPATCH * TLEN;
    mean_heads<<<(n + 255) / 256, 256, 0, stream>>>(w_buf, out_w);
  }
  mean_rows<<<(BB * DD + 255) / 256, 256, 0, stream>>>(out_attn, out_avg, NPATCH, 1.f / NPATCH);
  mean_rows<<<(BB * DD + 255) / 256, 256, 0, stream>>>(out_ptxt, out_cls, TLEN, 1.f / TLEN);
  cosine_scores<<<(BB * BB + 255) / 256, 256, 0, stream>>>(out_avg, out_cls, out_score);
}
